// RuleGNN_58463094833889
// MI455X (gfx1250) — matrix operand order repaired and checked by reading, not compiled, not run on hardware
//
#include <hip/hip_runtime.h>
#include <cstdint>
#include <cstddef>

// ---------------------------------------------------------------- constants
constexpr int N_ENT   = 10000;
constexpr int N_RELS  = 64;
constexpr int U_ACT   = 4;
constexpr int DD      = 128;
constexpr int NLAYER  = 2;
constexpr int E_PER_R = 4096;
constexpr int NEDGES  = N_RELS * E_PER_R;   // 262144
constexpr int NSEG    = N_RELS * N_ENT;     // 640000
constexpr int BATCH   = 64;
constexpr float LN_EPS = 1e-5f;
constexpr float SCALE  = 0.08838834764831845f; // 1/sqrt(128)
constexpr int MAT_ELEMS = 16384;            // 128x128

// ---------------------------------------------------------------- types
typedef __bf16 bf16_t;
typedef __attribute__((ext_vector_type(16))) __bf16 v16bf;
typedef __attribute__((ext_vector_type(8)))  __bf16 v8bf;
typedef __attribute__((ext_vector_type(8)))  float  v8f;
typedef __attribute__((ext_vector_type(4)))  unsigned v4u;
typedef __attribute__((ext_vector_type(8)))  int v8i;
typedef __attribute__((ext_vector_type(4)))  int v4i;

#if defined(__has_builtin)
#if __has_builtin(__builtin_amdgcn_tensor_load_to_lds)
#define HAVE_TDM 1
#endif
#endif
#ifndef HAVE_TDM
#define HAVE_TDM 0
#endif

__device__ inline bf16_t f2bf(float f) { return (bf16_t)f; }

__device__ inline v8f wmma_bf16(v16bf a, v16bf b, v8f c) {
  return __builtin_amdgcn_wmma_f32_16x16x32_bf16(false, a, false, b, (short)0, c,
                                                 false, false);
}

// A fragment (16x32 bf16, rows gathered): lane L holds row M=L&15;
// lanes<16: K = kc*32 + {0..15}; lanes>=16: +16.
__device__ inline v16bf load_a_frag(const bf16_t* __restrict__ row, int kc, int lane) {
  const int klo = (lane >> 4) << 4;
  const v8bf lo = *reinterpret_cast<const v8bf*>(row + (kc << 5) + klo);
  const v8bf hi = *reinterpret_cast<const v8bf*>(row + (kc << 5) + 8 + klo);
  v16bf a;
#pragma unroll
  for (int i = 0; i < 8; ++i) { a[i] = lo[i]; a[i + 8] = hi[i]; }
  return a;
}

// B fragment from (pre-swizzled) buffer: 16 contiguous bf16 per lane.
__device__ inline v16bf load_b_frag(const bf16_t* __restrict__ wsw, int nt, int kc,
                                    int lane) {
  return *reinterpret_cast<const v16bf*>(wsw + ((((nt << 2) + kc) << 9) + (lane << 4)));
}

// LDS byte offset of a __shared__ object (flat LDS addr low 32 bits)
__device__ inline unsigned lds_off(const void* p) {
  return (unsigned)(uintptr_t)p;
}

// TDM: DMA `nelem` bf16 elements from global to LDS (2D descriptor, 1 row tile).
__device__ inline void tdm_load(unsigned lds_addr, const void* gaddr, unsigned nelem) {
#if HAVE_TDM
  unsigned long long ga = (unsigned long long)gaddr;
  v4u g0;
  g0[0] = 1u;                                       // count=1 (valid), user mode
  g0[1] = lds_addr;                                 // lds_addr [63:32]
  g0[2] = (unsigned)(ga & 0xFFFFFFFFu);             // global_addr [95:64]
  g0[3] = (unsigned)((ga >> 32) & 0x01FFFFFFu) | (2u << 30);  // addr[56:32], type=2
  v8i g1;
  g1[0] = (int)(1u << 16);                          // wg_mask=0, data_size=1 (2B)
  g1[1] = (int)((nelem & 0xFFFFu) << 16);           // tensor_dim0[15:0] @ bits 63:48
  g1[2] = (int)(((nelem >> 16) & 0xFFFFu) | (1u << 16)); // dim0 hi, tensor_dim1=1
  g1[3] = (int)((nelem & 0xFFFFu) << 16);           // tile_dim0 @ bits 127:112
  g1[4] = 1;                                        // tile_dim1 = 1
  g1[5] = (int)nelem;                               // tensor_dim0_stride low32
  g1[6] = 0;
  g1[7] = 0;
  const v4i z4 = {0, 0, 0, 0};
  const v8i z8 = {0, 0, 0, 0, 0, 0, 0, 0};
  __builtin_amdgcn_tensor_load_to_lds(g0, g1, z4, z4, z8, 0);
#else
  (void)lds_addr; (void)gaddr; (void)nelem;
#endif
}

// cooperative fallback copy (also used when TDM unavailable)
__device__ inline void coop_copy(bf16_t* s, const bf16_t* g, int nelem, int tid,
                                 int nthr) {
  const v8bf* gv = reinterpret_cast<const v8bf*>(g);
  v8bf* sv = reinterpret_cast<v8bf*>(s);
  for (int i = tid; i < nelem / 8; i += nthr) sv[i] = gv[i];
}

// ordered-uint encoding of float for atomic max
__device__ inline unsigned ordf(float f) {
  unsigned u = __float_as_uint(f);
  return (u & 0x80000000u) ? ~u : (u | 0x80000000u);
}
__device__ inline float unordf(unsigned u) {
  return __uint_as_float((u & 0x80000000u) ? (u & 0x7fffffffu) : ~u);
}

// ---------------------------------------------------------------- prep kernels
__global__ __launch_bounds__(256) void k_zero(float* __restrict__ p, int n) {
  int i = blockIdx.x * 256 + threadIdx.x;
  if (i < n) p[i] = 0.f;
}

__global__ __launch_bounds__(256) void k_to_bf16(const float* __restrict__ src,
                                                 bf16_t* __restrict__ dst, int n) {
  int i = blockIdx.x * 256 + threadIdx.x;
  if (i < n) dst[i] = f2bf(src[i]);
}

// swizzle nmat contiguous 128x128 row-major (K x N) fp32 matrices into the
// WMMA B-fragment layout: k = kc*32 + (lane>>4)*16 + j, n = nt*16 + (lane&15)
__global__ __launch_bounds__(256) void k_swizzle(const float* __restrict__ src,
                                                 bf16_t* __restrict__ dst, int nmat) {
  int idx = blockIdx.x * 256 + threadIdx.x;
  if (idx >= nmat * MAT_ELEMS) return;
  int mat = idx >> 14;
  int rem = idx & 16383;
  int j    = rem & 15;
  int lane = (rem >> 4) & 31;
  int blk  = rem >> 9;         // nt*4 + kc
  int kc   = blk & 3;
  int nt   = blk >> 2;
  int n = (nt << 4) + (lane & 15);
  int k = (kc << 5) + ((lane >> 4) << 4) + j;
  dst[idx] = f2bf(src[(size_t)mat * MAT_ELEMS + (size_t)k * DD + n]);
}

// h_rel[l][r][d] = mean_o W_r[l][r][d][o]
__global__ __launch_bounds__(256) void k_hrel(const float* __restrict__ W_r,
                                              float* __restrict__ hrel) {
  int idx = blockIdx.x * 256 + threadIdx.x;   // 2*64*128
  if (idx >= NLAYER * N_RELS * DD) return;
  const float* p = W_r + (size_t)idx * DD;
  float s = 0.f;
#pragma unroll 8
  for (int o = 0; o < DD; ++o) s += p[o];
  hrel[idx] = s * (1.f / DD);
}

// hk2b[l][r][n] = h_rel[l][r] . Wk2[l][:,n] + bk[l][n]
__global__ __launch_bounds__(256) void k_hk2b(const float* __restrict__ hrel,
                                              const float* __restrict__ Wk,
                                              const float* __restrict__ bk,
                                              float* __restrict__ hk2b) {
  int idx = blockIdx.x * 256 + threadIdx.x;
  if (idx >= NLAYER * N_RELS * DD) return;
  int l = idx >> 13;
  int rem = idx & 8191;
  int r = rem >> 7;
  int n = rem & 127;
  const float* hv = hrel + ((size_t)l * N_RELS + r) * DD;
  const float* wv = Wk + ((size_t)l * 384 + 128) * DD + n;
  float s = 0.f;
#pragma unroll 8
  for (int d = 0; d < DD; ++d) s += hv[d] * wv[(size_t)d * DD];
  hk2b[idx] = s + bk[l * DD + n];
}

// ruleK[l][u][n] = rule_emb[l][rid[u]] . Wk3[l][:,n]
__global__ __launch_bounds__(256) void k_rulek(const float* __restrict__ rule_emb,
                                               const float* __restrict__ Wk,
                                               const int* __restrict__ rule_ids,
                                               float* __restrict__ ruleK) {
  int idx = blockIdx.x * 256 + threadIdx.x;
  if (idx >= NLAYER * U_ACT * DD) return;
  int l = idx >> 9;
  int rem = idx & 511;
  int u = rem >> 7;
  int n = rem & 127;
  int rid = rule_ids[u];
  const float* ev = rule_emb + ((size_t)l * 16 + rid) * DD;
  const float* wv = Wk + ((size_t)l * 384 + 256) * DD + n;
  float s = 0.f;
#pragma unroll 8
  for (int d = 0; d < DD; ++d) s += ev[d] * wv[(size_t)d * DD];
  ruleK[idx] = s;
}

// ---------------------------------------------------------------- scores (WMMA)
// one wave per 16-edge group; Wq/Wk1 staged to LDS once per block via TDM.
__global__ __launch_bounds__(256) void k_scores(
    const bf16_t* __restrict__ xb, const bf16_t* __restrict__ Wq_sw,
    const bf16_t* __restrict__ Wk1_sw, const float* __restrict__ bq,
    const float* __restrict__ hk2b, const float* __restrict__ ruleK,
    const int* __restrict__ src, const int* __restrict__ dst,
    const int* __restrict__ etype, float* __restrict__ sbuf) {
  __shared__ bf16_t sWq[MAT_ELEMS];
  __shared__ bf16_t sWk[MAT_ELEMS];
#if HAVE_TDM
  if (threadIdx.x < 32) {
    tdm_load(lds_off(sWq), Wq_sw, MAT_ELEMS);
    tdm_load(lds_off(sWk), Wk1_sw, MAT_ELEMS);
    __builtin_amdgcn_s_wait_tensorcnt(0);
  }
#else
  coop_copy(sWq, Wq_sw, MAT_ELEMS, threadIdx.x, 256);
  coop_copy(sWk, Wk1_sw, MAT_ELEMS, threadIdx.x, 256);
#endif
  __syncthreads();

  const int lane = threadIdx.x & 31;
  const int wid  = (blockIdx.x * 256 + threadIdx.x) >> 5;
  const int e0 = wid << 4;
  const int m  = lane & 15;
  const int hi = lane >> 4;
  const int drow = dst[e0 + m];
  const int srow = src[e0 + m];
  const int r = etype[e0];

  v16bf aq[4], ak[4];
#pragma unroll
  for (int kc = 0; kc < 4; ++kc) {
    aq[kc] = load_a_frag(xb + (size_t)drow * DD, kc, lane);
    ak[kc] = load_a_frag(xb + (size_t)srow * DD, kc, lane);
  }
  float pqb[8];
  float pqr[U_ACT][8];
#pragma unroll
  for (int v = 0; v < 8; ++v) {
    pqb[v] = 0.f;
#pragma unroll
    for (int u = 0; u < U_ACT; ++u) pqr[u][v] = 0.f;
  }

#pragma unroll
  for (int nt = 0; nt < 8; ++nt) {
    v8f qc = {};
    v8f kacc = {};
#pragma unroll
    for (int kc = 0; kc < 4; ++kc) {
      qc   = wmma_bf16(aq[kc], load_b_frag(sWq, nt, kc, lane), qc);
      kacc = wmma_bf16(ak[kc], load_b_frag(sWk, nt, kc, lane), kacc);
    }
    const int ncol = (nt << 4) + m;
    const float qbias = bq[ncol];
    const float kbias = hk2b[r * DD + ncol];
    float rk[U_ACT];
#pragma unroll
    for (int u = 0; u < U_ACT; ++u) rk[u] = ruleK[u * DD + ncol];
#pragma unroll
    for (int v = 0; v < 8; ++v) {
      const float qv = qc[v] + qbias;
      const float kv = kacc[v] + kbias;
      pqb[v] += qv * kv;
#pragma unroll
      for (int u = 0; u < U_ACT; ++u) pqr[u][v] += qv * rk[u];
    }
  }
  // butterfly reduce across the 16 lanes of each half-wave
#pragma unroll
  for (int v = 0; v < 8; ++v) {
#pragma unroll
    for (int msk = 1; msk < 16; msk <<= 1) {
      pqb[v] += __shfl_xor(pqb[v], msk, 16);
#pragma unroll
      for (int u = 0; u < U_ACT; ++u) pqr[u][v] += __shfl_xor(pqr[u][v], msk, 16);
    }
  }
  if (m < 8) {
    const int v = m;
    const int e = e0 + v + (hi << 3);
#pragma unroll
    for (int u = 0; u < U_ACT; ++u)
      sbuf[(size_t)u * NEDGES + e] = (pqb[v] + pqr[u][v]) * SCALE;
  }
}

// ---------------------------------------------------------------- softmax passes
__global__ __launch_bounds__(256) void k_segmax(const float* __restrict__ sbuf,
                                                const int* __restrict__ dst,
                                                const int* __restrict__ etype,
                                                unsigned* __restrict__ segmax) {
  int e = blockIdx.x * 256 + threadIdx.x;
  if (e >= NEDGES) return;
  int seg = etype[e] * N_ENT + dst[e];
#pragma unroll
  for (int u = 0; u < U_ACT; ++u)
    atomicMax(&segmax[(size_t)u * NSEG + seg], ordf(sbuf[(size_t)u * NEDGES + e]));
}

__global__ __launch_bounds__(256) void k_segexp(float* __restrict__ sbuf,
                                                const int* __restrict__ dst,
                                                const int* __restrict__ etype,
                                                const unsigned* __restrict__ segmax,
                                                float* __restrict__ segsum) {
  int e = blockIdx.x * 256 + threadIdx.x;
  if (e >= NEDGES) return;
  int seg = etype[e] * N_ENT + dst[e];
#pragma unroll
  for (int u = 0; u < U_ACT; ++u) {
    float mv = unordf(segmax[(size_t)u * NSEG + seg]);
    float ex = __expf(sbuf[(size_t)u * NEDGES + e] - mv);
    sbuf[(size_t)u * NEDGES + e] = ex;
    atomicAdd(&segsum[(size_t)u * NSEG + seg], ex);
  }
}

__global__ __launch_bounds__(256) void k_attnmean(const float* __restrict__ sbuf,
                                                  const int* __restrict__ dst,
                                                  const int* __restrict__ etype,
                                                  const float* __restrict__ segsum,
                                                  float* __restrict__ attnm) {
  int e = blockIdx.x * 256 + threadIdx.x;
  if (e >= NEDGES) return;
  int seg = etype[e] * N_ENT + dst[e];
  float s = 0.f;
#pragma unroll
  for (int u = 0; u < U_ACT; ++u)
    s += sbuf[(size_t)u * NEDGES + e] / segsum[(size_t)u * NSEG + seg];
  attnm[e] = 0.25f * s;
}

// ---------------------------------------------------------------- message (WMMA) + scatter
// 128 edges per block -> single relation per block; W_r[r] staged via TDM.
__global__ __launch_bounds__(256) void k_msg_scatter(
    const bf16_t* __restrict__ xb, const bf16_t* __restrict__ Wr_sw,
    const int* __restrict__ src, const int* __restrict__ dst,
    const int* __restrict__ etype, const float* __restrict__ attnm,
    float* __restrict__ outacc) {
  __shared__ bf16_t sW[MAT_ELEMS];
  const int r = __builtin_amdgcn_readfirstlane(etype[blockIdx.x * 128]);
  const bf16_t* wmat = Wr_sw + (size_t)r * MAT_ELEMS;
#if HAVE_TDM
  if (threadIdx.x < 32) {
    tdm_load(lds_off(sW), wmat, MAT_ELEMS);
    __builtin_amdgcn_s_wait_tensorcnt(0);
  }
#else
  coop_copy(sW, wmat, MAT_ELEMS, threadIdx.x, 256);
#endif
  __syncthreads();

  const int lane = threadIdx.x & 31;
  const int wid  = (blockIdx.x * 256 + threadIdx.x) >> 5;
  const int e0 = wid << 4;
  const int m  = lane & 15;
  const int hi = lane >> 4;
  const int srow = src[e0 + m];

  v16bf a[4];
#pragma unroll
  for (int kc = 0; kc < 4; ++kc) a[kc] = load_a_frag(xb + (size_t)srow * DD, kc, lane);

  float am[8];
  int dr[8];
#pragma unroll
  for (int v = 0; v < 8; ++v) {
    const int e = e0 + v + (hi << 3);
    am[v] = attnm[e];
    dr[v] = dst[e];
  }
#pragma unroll
  for (int nt = 0; nt < 8; ++nt) {
    v8f acc = {};
#pragma unroll
    for (int kc = 0; kc < 4; ++kc)
      acc = wmma_bf16(a[kc], load_b_frag(sW, nt, kc, lane), acc);
    const int ncol = (nt << 4) + m;
#pragma unroll
    for (int v = 0; v < 8; ++v)
      atomicAdd(&outacc[(size_t)dr[v] * DD + ncol], acc[v] * am[v]);
  }
}

// ---------------------------------------------------------------- bias + LN + ReLU
__global__ __launch_bounds__(256) void k_ln(const float* __restrict__ outacc,
                                            const float* __restrict__ conv_bias,
                                            const float* __restrict__ gamma,
                                            const float* __restrict__ beta,
                                            float* __restrict__ hcur,
                                            bf16_t* __restrict__ hbf) {
  const int lane = threadIdx.x & 31;
  const int row = (blockIdx.x * 256 + threadIdx.x) >> 5;
  if (row >= N_ENT) return;
  const int c0 = lane << 2;
  float4 x = *reinterpret_cast<const float4*>(outacc + (size_t)row * DD + c0);
  x.x += conv_bias[c0 + 0];
  x.y += conv_bias[c0 + 1];
  x.z += conv_bias[c0 + 2];
  x.w += conv_bias[c0 + 3];
  float s = x.x + x.y + x.z + x.w;
#pragma unroll
  for (int msk = 1; msk < 32; msk <<= 1) s += __shfl_xor(s, msk, 32);
  const float mu = s * (1.f / DD);
  float dx0 = x.x - mu, dx1 = x.y - mu, dx2 = x.z - mu, dx3 = x.w - mu;
  float q = dx0 * dx0 + dx1 * dx1 + dx2 * dx2 + dx3 * dx3;
#pragma unroll
  for (int msk = 1; msk < 32; msk <<= 1) q += __shfl_xor(q, msk, 32);
  const float inv = rsqrtf(q * (1.f / DD) + LN_EPS);
  float o[4] = {dx0, dx1, dx2, dx3};
#pragma unroll
  for (int i = 0; i < 4; ++i) {
    float v = o[i] * inv * gamma[c0 + i] + beta[c0 + i];
    v = fmaxf(v, 0.f);
    hcur[(size_t)row * DD + c0 + i] = v;
    hbf[(size_t)row * DD + c0 + i] = f2bf(v);
  }
}

// ---------------------------------------------------------------- final scorer
// hb = h @ W1b + b1 via WMMA, rows contiguous (10000 = 625 * 16)
__global__ __launch_bounds__(256) void k_hb(const bf16_t* __restrict__ hbf,
                                            const bf16_t* __restrict__ W1b_sw,
                                            const float* __restrict__ b1,
                                            float* __restrict__ hb) {
  const int lane = threadIdx.x & 31;
  const int wid  = (blockIdx.x * 256 + threadIdx.x) >> 5;
  if (wid >= N_ENT / 16) return;
  const int row0 = wid << 4;
  const int m  = lane & 15;
  const int hi = lane >> 4;
  v16bf a[4];
#pragma unroll
  for (int kc = 0; kc < 4; ++kc)
    a[kc] = load_a_frag(hbf + (size_t)(row0 + m) * DD, kc, lane);
#pragma unroll
  for (int nt = 0; nt < 8; ++nt) {
    v8f acc = {};
#pragma unroll
    for (int kc = 0; kc < 4; ++kc)
      acc = wmma_bf16(a[kc], load_b_frag(W1b_sw, nt, kc, lane), acc);
    const int ncol = (nt << 4) + m;
    const float bias = b1[ncol];
#pragma unroll
    for (int v = 0; v < 8; ++v) {
      const int rr = row0 + v + (hi << 3);
      hb[(size_t)rr * DD + ncol] = acc[v] + bias;
    }
  }
}

// ha0[b][n] = h[queries[b][0]] . W1a[:,n]   (tiny: 64x128)
__global__ __launch_bounds__(256) void k_ha(const float* __restrict__ hcur,
                                            const float* __restrict__ W1,
                                            const int* __restrict__ queries,
                                            float* __restrict__ ha0) {
  int idx = blockIdx.x * 256 + threadIdx.x;
  if (idx >= BATCH * DD) return;
  int b = idx >> 7;
  int n = idx & 127;
  int qrow = queries[2 * b];
  const float* hv = hcur + (size_t)qrow * DD;
  const float* wv = W1 + n;  // W1a = W1[:128]
  float s = 0.f;
#pragma unroll 8
  for (int k = 0; k < DD; ++k) s += hv[k] * wv[(size_t)k * DD];
  ha0[idx] = s;
}

// out[b][n] = sum_k relu(ha0[b][k] + hb[n][k]) * W2[k] + b2
__global__ __launch_bounds__(256) void k_final(const float* __restrict__ ha0,
                                               const float* __restrict__ hb,
                                               const float* __restrict__ W2,
                                               const float* __restrict__ b2,
                                               float* __restrict__ out) {
  int n = blockIdx.x * 256 + threadIdx.x;
  int b = blockIdx.y;
  if (n >= N_ENT) return;
  const float* ha = ha0 + (size_t)b * DD;
  const float* hv = hb + (size_t)n * DD;
  float acc = 0.f;
#pragma unroll 8
  for (int k = 0; k < DD; ++k) {
    float z = fmaxf(ha[k] + hv[k], 0.f);
    acc += z * W2[k];
  }
  out[(size_t)b * N_ENT + n] = acc + b2[0];
}

// ---------------------------------------------------------------- host
static inline int div_up(int a, int b) { return (a + b - 1) / b; }

extern "C" void kernel_launch(void* const* d_in, const int* in_sizes, int n_in,
                              void* d_out, int out_size, void* d_ws, size_t ws_size,
                              hipStream_t stream) {
  (void)in_sizes; (void)n_in; (void)out_size; (void)ws_size;
  const float* entity   = (const float*)d_in[0];
  const float* W_r      = (const float*)d_in[1];
  const float* Wq       = (const float*)d_in[2];
  const float* bq       = (const float*)d_in[3];
  const float* Wk       = (const float*)d_in[4];
  const float* bk       = (const float*)d_in[5];
  const float* rule_emb = (const float*)d_in[6];
  const float* conv_b   = (const float*)d_in[7];
  const float* ln_g     = (const float*)d_in[8];
  const float* ln_b     = (const float*)d_in[9];
  const float* W1       = (const float*)d_in[10];
  const float* b1       = (const float*)d_in[11];
  const float* W2       = (const float*)d_in[12];
  const float* b2       = (const float*)d_in[13];
  const int*   queries  = (const int*)d_in[14];
  const int*   edge_idx = (const int*)d_in[15];
  const int*   etype    = (const int*)d_in[16];
  const int*   rule_ids = (const int*)d_in[17];
  const int* src = edge_idx;
  const int* dst = edge_idx + NEDGES;
  float* out = (float*)d_out;

  // ---- workspace layout
  size_t off = 0;
  auto alloc = [&](size_t bytes) -> void* {
    off = (off + 255) & ~(size_t)255;
    void* p = (char*)d_ws + off;
    off += bytes;
    return p;
  };
  bf16_t*   xb0    = (bf16_t*)alloc((size_t)N_ENT * DD * 2);
  bf16_t*   hbf    = (bf16_t*)alloc((size_t)N_ENT * DD * 2);
  float*    hcur   = (float*)alloc((size_t)N_ENT * DD * 4);
  float*    outacc = (float*)alloc((size_t)N_ENT * DD * 4);
  bf16_t*   Wq_sw  = (bf16_t*)alloc((size_t)NLAYER * MAT_ELEMS * 2);
  bf16_t*   Wk1_sw = (bf16_t*)alloc((size_t)NLAYER * MAT_ELEMS * 2);
  bf16_t*   Wr_sw  = (bf16_t*)alloc((size_t)NLAYER * N_RELS * MAT_ELEMS * 2);
  bf16_t*   W1b_sw = (bf16_t*)alloc((size_t)MAT_ELEMS * 2);
  float*    hrel   = (float*)alloc((size_t)NLAYER * N_RELS * DD * 4);
  float*    hk2b   = (float*)alloc((size_t)NLAYER * N_RELS * DD * 4);
  float*    ruleK  = (float*)alloc((size_t)NLAYER * U_ACT * DD * 4);
  float*    sbuf   = (float*)alloc((size_t)U_ACT * NEDGES * 4);
  unsigned* segmax = (unsigned*)alloc((size_t)U_ACT * NSEG * 4);
  float*    segsum = (float*)alloc((size_t)U_ACT * NSEG * 4);
  float*    attnm  = (float*)alloc((size_t)NEDGES * 4);
  float*    hb     = (float*)alloc((size_t)N_ENT * DD * 4);
  float*    ha0    = (float*)alloc((size_t)BATCH * DD * 4);

  // ---- prep
  k_to_bf16<<<div_up(N_ENT * DD, 256), 256, 0, stream>>>(entity, xb0, N_ENT * DD);
  for (int l = 0; l < NLAYER; ++l) {
    k_swizzle<<<64, 256, 0, stream>>>(Wq + (size_t)l * MAT_ELEMS,
                                      Wq_sw + (size_t)l * MAT_ELEMS, 1);
    k_swizzle<<<64, 256, 0, stream>>>(Wk + (size_t)l * 384 * DD,
                                      Wk1_sw + (size_t)l * MAT_ELEMS, 1);
    k_swizzle<<<4096, 256, 0, stream>>>(W_r + (size_t)l * N_RELS * MAT_ELEMS,
                                        Wr_sw + (size_t)l * N_RELS * MAT_ELEMS, N_RELS);
  }
  k_swizzle<<<64, 256, 0, stream>>>(W1 + MAT_ELEMS, W1b_sw, 1);
  k_hrel<<<div_up(NLAYER * N_RELS * DD, 256), 256, 0, stream>>>(W_r, hrel);
  k_hk2b<<<div_up(NLAYER * N_RELS * DD, 256), 256, 0, stream>>>(hrel, Wk, bk, hk2b);
  k_rulek<<<div_up(NLAYER * U_ACT * DD, 256), 256, 0, stream>>>(rule_emb, Wk, rule_ids,
                                                                ruleK);

  // ---- layers
  const bf16_t* xin = xb0;
  const int egrids = div_up(NEDGES, 256);
  const int wgrids = (NEDGES / 16) / 8;  // 16-edge groups, 8 waves per block
  for (int l = 0; l < NLAYER; ++l) {
    k_zero<<<div_up(N_ENT * DD, 256), 256, 0, stream>>>(outacc, N_ENT * DD);
    k_zero<<<div_up(U_ACT * NSEG, 256), 256, 0, stream>>>((float*)segmax, U_ACT * NSEG);
    k_zero<<<div_up(U_ACT * NSEG, 256), 256, 0, stream>>>(segsum, U_ACT * NSEG);

    k_scores<<<wgrids, 256, 0, stream>>>(xin, Wq_sw + (size_t)l * MAT_ELEMS,
                                         Wk1_sw + (size_t)l * MAT_ELEMS, bq + l * DD,
                                         hk2b + (size_t)l * N_RELS * DD,
                                         ruleK + (size_t)l * U_ACT * DD, src, dst,
                                         etype, sbuf);
    k_segmax<<<egrids, 256, 0, stream>>>(sbuf, dst, etype, segmax);
    k_segexp<<<egrids, 256, 0, stream>>>(sbuf, dst, etype, segmax, segsum);
    k_attnmean<<<egrids, 256, 0, stream>>>(sbuf, dst, etype, segsum, attnm);
    k_msg_scatter<<<wgrids, 256, 0, stream>>>(xin,
                                              Wr_sw + (size_t)l * N_RELS * MAT_ELEMS,
                                              src, dst, etype, attnm, outacc);
    k_ln<<<div_up(N_ENT * 32, 256), 256, 0, stream>>>(outacc, conv_b + l * DD,
                                                      ln_g + l * DD, ln_b + l * DD,
                                                      hcur, hbf);
    xin = hbf;
  }

  // ---- final scorer
  k_hb<<<div_up((N_ENT / 16) * 32, 256), 256, 0, stream>>>(hbf, W1b_sw, b1, hb);
  k_ha<<<div_up(BATCH * DD, 256), 256, 0, stream>>>(hcur, W1, queries, ha0);
  dim3 fgrid(div_up(N_ENT, 256), BATCH);
  k_final<<<fgrid, 256, 0, stream>>>(ha0, hb, W2, b2, out);
}
